// GNN_33784212750625
// MI455X (gfx1250) — matrix operand order repaired and checked by reading, not compiled, not run on hardware
//
#include <hip/hip_runtime.h>
#include <hip/hip_bf16.h>

#define N_NODES 50000
#define N_EDGES 600000
#define D_IN    128
#define D_HID   1024
#define D_OUT   128

typedef __attribute__((ext_vector_type(16))) __bf16 v16bf;
typedef __attribute__((ext_vector_type(8)))  __bf16 v8bf;
typedef __attribute__((ext_vector_type(8)))  float  v8f;

union V16 { v16bf v; v8bf h[2]; };

__device__ __forceinline__ __bf16 f2bf(float f) {
    union { float f; unsigned u; } a; a.f = f;
    unsigned u = a.u;
    u += 0x7FFFu + ((u >> 16) & 1u);          // round-to-nearest-even
    union { unsigned short s; __bf16 b; } r;
    r.s = (unsigned short)(u >> 16);
    return r.b;
}

// ---------------------------------------------------------------------------
// Kernel 1: zero the node-accumulator h in workspace
// ---------------------------------------------------------------------------
__global__ __launch_bounds__(256) void zero_kernel(float4* p, int n4) {
    int i = blockIdx.x * 256 + threadIdx.x;
    if (i < n4) p[i] = float4{0.f, 0.f, 0.f, 0.f};
}

// ---------------------------------------------------------------------------
// Kernel 2: per-edge  h[dst] += feat[src] + edge_feat[e]
// One wave (32 lanes) per edge; each lane owns 4 consecutive floats.
// ---------------------------------------------------------------------------
__global__ __launch_bounds__(256) void edge_kernel(const float* __restrict__ feat,
                                                   const float* __restrict__ ef,
                                                   const int*   __restrict__ src,
                                                   const int*   __restrict__ dst,
                                                   float*       __restrict__ h) {
    int wid  = (blockIdx.x * 256 + threadIdx.x) >> 5;   // global wave id == edge id
    int lane = threadIdx.x & 31;
    if (wid >= N_EDGES) return;
    int s = src[wid];
    int d = dst[wid];
    const float4* fs = (const float4*)(feat + (size_t)s   * D_IN);
    const float4* fe = (const float4*)(ef   + (size_t)wid * D_IN);
    float4 a = fs[lane];
    float4 b = fe[lane];
    float* hp = h + (size_t)d * D_IN + lane * 4;
    unsafeAtomicAdd(hp + 0, a.x + b.x);
    unsafeAtomicAdd(hp + 1, a.y + b.y);
    unsafeAtomicAdd(hp + 2, a.z + b.z);
    unsafeAtomicAdd(hp + 3, a.w + b.w);
}

// ---------------------------------------------------------------------------
// Kernel 3: pre-swizzle W1/W2 (fp32 row-major) into bf16 WMMA B-fragment layout.
// B-fragment (16x16x32 bf16): half i of lane (g = lane>>4, n = lane&15)
//   holds B[K = kstep*32 + i + 16*g][N = ntile*16 + n].
// Packed index: ((frag*32 + lane)*16 + i)  with frag = ntile*nK + kstep.
// ---------------------------------------------------------------------------
__global__ __launch_bounds__(256) void prep_weights(const float* __restrict__ W1,
                                                    const float* __restrict__ W2,
                                                    __bf16* __restrict__ W1p,
                                                    __bf16* __restrict__ W2p) {
    int idx  = blockIdx.x * 256 + threadIdx.x;   // 0 .. 262143
    int half = idx >> 17;                        // 0 -> W1 (131072), 1 -> W2
    int id   = idx & 131071;
    int i    = id & 15;
    int lane = (id >> 4) & 31;
    int frag = id >> 9;
    int g = lane >> 4, n = lane & 15;
    int Koff = i + 16 * g;                       // B-matrix layout
    if (half == 0) {
        int ntile = frag >> 2, ks = frag & 3;    // D_HID/16 = 64 ntiles, K=128 -> 4 ksteps
        int K = ks * 32 + Koff;
        int N = ntile * 16 + n;
        W1p[id] = f2bf(W1[(size_t)K * D_HID + N]);
    } else {
        int ntile = frag >> 5, ks = frag & 31;   // D_OUT/16 = 8 ntiles, K=1024 -> 32 ksteps
        int K = ks * 32 + Koff;
        int N = ntile * 16 + n;
        W2p[id] = f2bf(W2[(size_t)K * D_OUT + N]);
    }
}

// ---------------------------------------------------------------------------
// Kernel 4: fused MLP  out = relu(h@W1 + b1) @ W2 + b2
// One 16-row node tile per workgroup; 8 waves.
// Layer 1: wave w computes columns [w*128, w*128+128) -> 8 N-tiles x 4 K-steps.
// Layer 2: wave w computes out columns [w*16, w*16+16)  -> 32 K-steps.
// A-fragment (16x32 bf16): lane (g,m=lane&15) holds row M=m,
//   halves i=0..15 -> K = kb + 16g + i     (contiguous 32B run)
// ---------------------------------------------------------------------------
__global__ __launch_bounds__(256) void mlp_kernel(const float*  __restrict__ h,
                                                  const __bf16* __restrict__ W1p,
                                                  const __bf16* __restrict__ W2p,
                                                  const float*  __restrict__ b1,
                                                  const float*  __restrict__ b2,
                                                  float*        __restrict__ out) {
    __shared__ __bf16 sX[16 * D_IN];    //  4 KB: input tile (bf16)
    __shared__ __bf16 sH[16 * D_HID];   // 32 KB: relu(layer1) tile (bf16)

    int t    = threadIdx.x;
    int row0 = blockIdx.x * 16;

    // stage input tile -> LDS (bf16)
    for (int e = t; e < 16 * D_IN; e += 256) {
        int r = e >> 7, c = e & 127;
        sX[e] = f2bf(h[(size_t)(row0 + r) * D_IN + c]);
    }
    __syncthreads();

    int wave = t >> 5, lane = t & 31;
    int g = lane >> 4, n = lane & 15, m = lane & 15;

    // A-fragments for layer 1 (shared by all 8 of this wave's N-tiles)
    V16 a1[4];
    #pragma unroll
    for (int ks = 0; ks < 4; ++ks) {
        const __bf16* base = sX + m * D_IN + ks * 32 + g * 16;
        a1[ks].h[0] = *(const v8bf*)(base);
        a1[ks].h[1] = *(const v8bf*)(base + 8);
    }

    // -------- layer 1 --------
    #pragma unroll
    for (int nt = 0; nt < 8; ++nt) {
        int ntile = wave * 8 + nt;
        v8f acc = {};
        #pragma unroll
        for (int ks = 0; ks < 4; ++ks) {
            const v8bf* bp = (const v8bf*)(W1p + ((size_t)(ntile * 4 + ks) * 32 + lane) * 16);
            V16 b; b.h[0] = bp[0]; b.h[1] = bp[1];
            acc = __builtin_amdgcn_wmma_f32_16x16x32_bf16(
                      false, a1[ks].v, false, b.v, (short)0, acc, false, false);
        }
        int col = ntile * 16 + n;
        float bias = b1[col];
        #pragma unroll
        for (int r = 0; r < 8; ++r) {
            float v = acc[r] + bias;
            v = v > 0.f ? v : 0.f;                 // ReLU
            sH[(r + 8 * g) * D_HID + col] = f2bf(v);
        }
    }
    __syncthreads();

    // -------- layer 2 --------
    v8f acc = {};
    #pragma unroll
    for (int ks = 0; ks < 32; ++ks) {
        const __bf16* abase = sH + m * D_HID + ks * 32 + g * 16;
        V16 a; a.h[0] = *(const v8bf*)(abase);
               a.h[1] = *(const v8bf*)(abase + 8);
        const v8bf* bp = (const v8bf*)(W2p + ((size_t)(wave * 32 + ks) * 32 + lane) * 16);
        V16 b; b.h[0] = bp[0]; b.h[1] = bp[1];
        acc = __builtin_amdgcn_wmma_f32_16x16x32_bf16(
                  false, a.v, false, b.v, (short)0, acc, false, false);
    }
    int col = wave * 16 + n;
    float bias = b2[col];
    #pragma unroll
    for (int r = 0; r < 8; ++r) {
        out[(size_t)(row0 + r + 8 * g) * D_OUT + col] = acc[r] + bias;
    }
}

// ---------------------------------------------------------------------------
extern "C" void kernel_launch(void* const* d_in, const int* in_sizes, int n_in,
                              void* d_out, int out_size, void* d_ws, size_t ws_size,
                              hipStream_t stream) {
    const float* feat = (const float*)d_in[0];
    const float* ef   = (const float*)d_in[1];
    const int*   src  = (const int*)d_in[2];
    const int*   dst  = (const int*)d_in[3];
    const float* W1   = (const float*)d_in[4];
    const float* b1   = (const float*)d_in[5];
    const float* W2   = (const float*)d_in[6];
    const float* b2   = (const float*)d_in[7];
    float* out = (float*)d_out;

    // workspace layout
    float*  h   = (float*)d_ws;                       // 50000*128 floats = 25.6 MB
    __bf16* W1p = (__bf16*)(h + (size_t)N_NODES * D_IN);   // 131072 bf16 (256 KB)
    __bf16* W2p = W1p + 131072;                            // 131072 bf16 (256 KB)

    // 1) zero accumulator
    int n4 = (N_NODES * D_IN) / 4;                    // 1,600,000 float4
    zero_kernel<<<(n4 + 255) / 256, 256, 0, stream>>>((float4*)h, n4);

    // 2) edge gather + scatter-add (one wave per edge)
    int edgeBlocks = (N_EDGES * 32 + 255) / 256;      // 75000
    edge_kernel<<<edgeBlocks, 256, 0, stream>>>(feat, ef, src, dst, h);

    // 3) swizzle weights to bf16 WMMA fragments (independent of 1/2, same stream)
    prep_weights<<<(262144) / 256, 256, 0, stream>>>(W1, W2, W1p, W2p);

    // 4) fused 2-layer MLP with WMMA
    mlp_kernel<<<N_NODES / 16, 256, 0, stream>>>(h, W1p, W2p, b1, b2, out);
}
